// Graph_conv_11381663334633
// MI455X (gfx1250) — matrix operand order repaired and checked by reading, not compiled, not run on hardware
//
#include <hip/hip_runtime.h>
#include <hip/hip_bf16.h>
#include <math.h>

// ---------------------------------------------------------------------------
// Types for CDNA5 WMMA (wave32, 16x16x32 bf16 -> f32)
// ---------------------------------------------------------------------------
typedef __attribute__((ext_vector_type(16))) __bf16        v16bf;
typedef __attribute__((ext_vector_type(8)))  float         v8f;
typedef __attribute__((ext_vector_type(4)))  unsigned int  u32x4;
typedef __hip_bfloat16 bf16_t;

#define NPIX  2704   // 52*52
#define NPAD  2720   // NPIX padded to multiple of 32 (K padding)
#define DD    1024
#define CHID  512
#define KCONV 9216   // 1024*3*3
#define IMGW  52

static __device__ __forceinline__ bf16_t f2b(float x) { return __float2bfloat16(x); }

// ---------------------------------------------------------------------------
// Register-blocked WMMA GEMM:  C[M][ldc] = A[M][KP] * B^T  (B staged [Nc][KP])
// Each wave computes a 32x64 tile = 2x4 WMMA tiles (8 accumulators).
// Per K-step(32): 2 A-fragments + 4 B-fragments feed 8 WMMAs
//   -> 1.5 b128 loads per WMMA (vs 4 for unblocked).
// Edge tiles clamp load rows/cols (branch-free; EXEC stays all-ones for WMMA)
// and bounds-check stores. KP must be a multiple of 32.
// Epilogue: optional C-init accumulate, per-col/per-row bias, ReLU.
// ---------------------------------------------------------------------------
__global__ __launch_bounds__(256) void k_gemm(
    const bf16_t* __restrict__ A, const bf16_t* __restrict__ B,
    float* __restrict__ C, const float* __restrict__ Cinit,
    const float* __restrict__ bias,
    int M, int Nc, int KP, int ldc, int biasMode, int relu)
{
    const int wave = threadIdx.x >> 5;
    const int lane = threadIdx.x & 31;
    const int mb = (M + 31) >> 5;                  // 32-row wave tiles
    const int nb = (Nc + 63) >> 6;                 // 64-col wave tiles
    const int tile = blockIdx.x * 8 + wave;        // consecutive tiles share A rows
    if (tile >= mb * nb) return;                   // wave-uniform exit
    const int tm   = tile / nb;
    const int tn   = tile - tm * nb;
    const int half = lane >> 4;
    const int l15  = lane & 15;

    // Clamped source rows/cols (duplicates computed for OOB, never stored)
    int r0 = tm * 32 + l15;       if (r0 > M - 1) r0 = M - 1;
    int r1 = tm * 32 + 16 + l15;  if (r1 > M - 1) r1 = M - 1;
    const bf16_t* arow0 = A + (size_t)r0 * KP;
    const bf16_t* arow1 = A + (size_t)r1 * KP;
    const bf16_t* brow[4];
#pragma unroll
    for (int j = 0; j < 4; ++j) {
        int c = tn * 64 + j * 16 + l15;
        if (c > Nc - 1) c = Nc - 1;
        brow[j] = B + (size_t)c * KP;
    }

    v8f acc[2][4];
#pragma unroll
    for (int i = 0; i < 2; ++i)
#pragma unroll
        for (int j = 0; j < 4; ++j)
            acc[i][j] = (v8f){0.f, 0.f, 0.f, 0.f, 0.f, 0.f, 0.f, 0.f};

    union FragA { v16bf v; u32x4 q[2]; };

    for (int k0 = 0; k0 < KP; k0 += 32) {
        // A fragments (16x32 layout): k = 16*half + j (contiguous, matches B)
        FragA fa0, fa1;
        fa0.q[0] = *(const u32x4*)(arow0 + k0 + 16 * half);
        fa0.q[1] = *(const u32x4*)(arow0 + k0 + 16 * half + 8);
        fa1.q[0] = *(const u32x4*)(arow1 + k0 + 16 * half);
        fa1.q[1] = *(const u32x4*)(arow1 + k0 + 16 * half + 8);
        // B fragments (32x16 layout): lane holds col, k = 16*half + j (contiguous 32B)
        v16bf fb0 = *(const v16bf*)(brow[0] + k0 + 16 * half);
        v16bf fb1 = *(const v16bf*)(brow[1] + k0 + 16 * half);
        v16bf fb2 = *(const v16bf*)(brow[2] + k0 + 16 * half);
        v16bf fb3 = *(const v16bf*)(brow[3] + k0 + 16 * half);
        acc[0][0] = __builtin_amdgcn_wmma_f32_16x16x32_bf16(false, fa0.v, false, fb0, (short)0, acc[0][0], false, false);
        acc[0][1] = __builtin_amdgcn_wmma_f32_16x16x32_bf16(false, fa0.v, false, fb1, (short)0, acc[0][1], false, false);
        acc[0][2] = __builtin_amdgcn_wmma_f32_16x16x32_bf16(false, fa0.v, false, fb2, (short)0, acc[0][2], false, false);
        acc[0][3] = __builtin_amdgcn_wmma_f32_16x16x32_bf16(false, fa0.v, false, fb3, (short)0, acc[0][3], false, false);
        acc[1][0] = __builtin_amdgcn_wmma_f32_16x16x32_bf16(false, fa1.v, false, fb0, (short)0, acc[1][0], false, false);
        acc[1][1] = __builtin_amdgcn_wmma_f32_16x16x32_bf16(false, fa1.v, false, fb1, (short)0, acc[1][1], false, false);
        acc[1][2] = __builtin_amdgcn_wmma_f32_16x16x32_bf16(false, fa1.v, false, fb2, (short)0, acc[1][2], false, false);
        acc[1][3] = __builtin_amdgcn_wmma_f32_16x16x32_bf16(false, fa1.v, false, fb3, (short)0, acc[1][3], false, false);
    }

    // Epilogue (divergence allowed here; all WMMAs are done)
#pragma unroll
    for (int i = 0; i < 2; ++i) {
#pragma unroll
        for (int j = 0; j < 4; ++j) {
            const int n = tn * 64 + j * 16 + l15;
            if (n >= Nc) continue;
            float bcol = (biasMode == 1) ? bias[n] : 0.f;
#pragma unroll
            for (int r = 0; r < 8; ++r) {
                const int m = tm * 32 + i * 16 + half * 8 + r;  // C: VGPR r -> M=r+8*half
                if (m >= M) continue;
                float v = acc[i][j][r];
                if (Cinit) v += Cinit[(size_t)m * ldc + n];
                if (biasMode == 1) v += bcol;
                else if (biasMode == 2) v += bias[m];
                if (relu) v = fmaxf(v, 0.f);
                C[(size_t)m * ldc + n] = v;
            }
        }
    }
}

// ---------------------------------------------------------------------------
// Staging kernels: f32 -> bf16 in WMMA operand layouts (zero K-padding)
// ---------------------------------------------------------------------------
__global__ void k_stageA(const float* __restrict__ src, bf16_t* __restrict__ dst,
                         int M, int K, int lds, int KP)
{
    int i = blockIdx.x * blockDim.x + threadIdx.x;
    if (i >= M * KP) return;
    int m = i / KP, k = i - m * KP;
    dst[i] = f2b((k < K) ? src[(size_t)m * lds + k] : 0.f);
}

__global__ void k_stageB(const float* __restrict__ src, bf16_t* __restrict__ dst,
                         int K, int Nc, int lds, int KP)   // dst[n][k] = src[k][n]
{
    int i = blockIdx.x * blockDim.x + threadIdx.x;
    if (i >= Nc * KP) return;
    int n = i / KP, k = i - n * KP;
    dst[i] = f2b((k < K) ? src[(size_t)k * lds + n] : 0.f);
}

__global__ void k_stageSymB(const float* __restrict__ src, bf16_t* __restrict__ dst, int D)
{   // B-operand of Asym = 0.5*(A + A^T)
    int i = blockIdx.x * blockDim.x + threadIdx.x;
    if (i >= D * D) return;
    int n = i / D, k = i - n * D;
    dst[i] = f2b(0.5f * (src[k * D + n] + src[n * D + k]));
}

// ---------------------------------------------------------------------------
// Bilinear mask resize 416x416 -> 52x52 (align-corners linspace, as reference)
// ---------------------------------------------------------------------------
__global__ void k_resize_mask(const float* __restrict__ mask, float* __restrict__ pm)
{
    int i = blockIdx.x * blockDim.x + threadIdx.x;
    if (i >= NPIX) return;
    int oy = i / IMGW, ox = i % IMGW;
    const float sc = 415.0f / 51.0f;
    float ys = oy * sc, xs = ox * sc;
    int y0 = (int)floorf(ys), x0 = (int)floorf(xs);
    int y1 = (y0 + 1 < 415) ? y0 + 1 : 415;
    int x1 = (x0 + 1 < 415) ? x0 + 1 : 415;
    float wy = ys - (float)y0, wx = xs - (float)x0;
    float Ia = mask[y0 * 416 + x0], Ib = mask[y0 * 416 + x1];
    float Ic = mask[y1 * 416 + x0], Id = mask[y1 * 416 + x1];
    pm[i] = Ia * (1.f - wy) * (1.f - wx) + Ib * (1.f - wy) * wx
          + Ic * wy * (1.f - wx) + Id * wy * wx;
}

// ---------------------------------------------------------------------------
// Fused im2col (3x3, pad 1) + mask-multiply (batch 0), written transposed as
// B-operand: dst[p][k], k = ci*9 + kh*3 + kw, bf16.
// ---------------------------------------------------------------------------
__global__ void k_im2colT(const float* __restrict__ x, const float* __restrict__ pm,
                          bf16_t* __restrict__ dst, int batch)
{
    int idx = blockIdx.x * blockDim.x + threadIdx.x;
    if (idx >= NPIX * 1024) return;
    int p  = idx >> 10;
    int ci = idx & 1023;
    int oh = p / IMGW, ow = p % IMGW;
    const float* xb = x + ((size_t)batch * 1024 + ci) * NPIX;
    bf16_t* d = dst + (size_t)p * KCONV + ci * 9;
#pragma unroll
    for (int kh = 0; kh < 3; ++kh)
#pragma unroll
        for (int kw = 0; kw < 3; ++kw) {
            int ih = oh + kh - 1, iw = ow + kw - 1;
            float v = 0.f;
            if (ih >= 0 && ih < IMGW && iw >= 0 && iw < IMGW) {
                v = xb[ih * IMGW + iw];
                if (batch == 0) v *= pm[ih * IMGW + iw];
            }
            d[kh * 3 + kw] = f2b(v);
        }
}

// ---------------------------------------------------------------------------
// Column-wise softmax over rows (axis=1 of (1,N,N)): coalesced thread-per-col
// ---------------------------------------------------------------------------
__global__ void k_softmax_col(float* __restrict__ A, int M, int Nc)
{
    int n = blockIdx.x * blockDim.x + threadIdx.x;
    if (n >= Nc) return;
    float mx = -__builtin_inff();
    for (int m = 0; m < M; ++m) mx = fmaxf(mx, A[(size_t)m * Nc + n]);
    float s = 0.f;
    for (int m = 0; m < M; ++m) s += expf(A[(size_t)m * Nc + n] - mx);
    float inv = 1.f / s;
    for (int m = 0; m < M; ++m) {
        size_t o = (size_t)m * Nc + n;
        A[o] = expf(A[o] - mx) * inv;
    }
}

// ---------------------------------------------------------------------------
// Global min/max reduction + in-place (x-min)/(max-min)
// ---------------------------------------------------------------------------
__global__ void k_minmax_partial(const float* __restrict__ x, size_t n, float* __restrict__ red)
{
    __shared__ float smn[256], smx[256];
    float mn = __builtin_inff(), mx = -__builtin_inff();
    for (size_t i = (size_t)blockIdx.x * blockDim.x + threadIdx.x; i < n;
         i += (size_t)gridDim.x * blockDim.x) {
        float v = x[i];
        mn = fminf(mn, v); mx = fmaxf(mx, v);
    }
    smn[threadIdx.x] = mn; smx[threadIdx.x] = mx;
    __syncthreads();
    for (int s = 128; s > 0; s >>= 1) {
        if (threadIdx.x < s) {
            smn[threadIdx.x] = fminf(smn[threadIdx.x], smn[threadIdx.x + s]);
            smx[threadIdx.x] = fmaxf(smx[threadIdx.x], smx[threadIdx.x + s]);
        }
        __syncthreads();
    }
    if (threadIdx.x == 0) { red[blockIdx.x] = smn[0]; red[gridDim.x + blockIdx.x] = smx[0]; }
}

__global__ void k_minmax_final(float* __restrict__ red, int nb)
{
    __shared__ float smn[256], smx[256];
    float mn = __builtin_inff(), mx = -__builtin_inff();
    for (int i = threadIdx.x; i < nb; i += 256) {
        mn = fminf(mn, red[i]); mx = fmaxf(mx, red[nb + i]);
    }
    smn[threadIdx.x] = mn; smx[threadIdx.x] = mx;
    __syncthreads();
    for (int s = 128; s > 0; s >>= 1) {
        if (threadIdx.x < s) {
            smn[threadIdx.x] = fminf(smn[threadIdx.x], smn[threadIdx.x + s]);
            smx[threadIdx.x] = fmaxf(smx[threadIdx.x], smx[threadIdx.x + s]);
        }
        __syncthreads();
    }
    if (threadIdx.x == 0) { red[0] = smn[0]; red[1] = smx[0]; }
}

__global__ void k_norm(float* __restrict__ x, size_t n, const float* __restrict__ red)
{
    float mn = red[0], inv = 1.f / (red[1] - red[0]);
    for (size_t i = (size_t)blockIdx.x * blockDim.x + threadIdx.x; i < n;
         i += (size_t)gridDim.x * blockDim.x)
        x[i] = (x[i] - mn) * inv;
}

// ---------------------------------------------------------------------------
// Host orchestration
// ---------------------------------------------------------------------------
static inline int cdiv(int a, int b) { return (a + b - 1) / b; }

extern "C" void kernel_launch(void* const* d_in, const int* in_sizes, int n_in,
                              void* d_out, int out_size, void* d_ws, size_t ws_size,
                              hipStream_t stream)
{
    (void)in_sizes; (void)n_in; (void)out_size;
    const float* x       = (const float*)d_in[0];
    const float* maskp   = (const float*)d_in[1];
    const float* conv_w  = (const float*)d_in[2];
    const float* conv_b  = (const float*)d_in[3];
    const float* a_w0    = (const float*)d_in[4];
    const float* a_b0    = (const float*)d_in[5];
    const float* u_w0    = (const float*)d_in[6];
    const float* u_b0    = (const float*)d_in[7];
    const float* a_w1    = (const float*)d_in[8];
    const float* a_b1    = (const float*)d_in[9];
    const float* u_w1    = (const float*)d_in[10];
    const float* u_b1    = (const float*)d_in[11];
    const float* aff0    = (const float*)d_in[12];
    const float* aff1    = (const float*)d_in[13];
    const float* cross_w = (const float*)d_in[14];
    const float* cross_b = (const float*)d_in[15];

    // ---- workspace bump allocation (deterministic layout) ----
    char* base = (char*)d_ws;
    size_t off = 0;
    auto alloc = [&](size_t bytes) -> char* {
        char* p = base + off;
        off += (bytes + 255) & ~(size_t)255;
        return p;
    };
    float*  RED   = (float*) alloc(2048 * 4);
    float*  PM    = (float*) alloc((size_t)NPIX * 4);
    float*  FEAT  = (float*) alloc((size_t)2 * CHID * NPIX * 4);  // conv out == x_s/x_q
    bf16_t* XS16  = (bf16_t*)alloc((size_t)NPIX * CHID * 2);
    bf16_t* XQ16  = (bf16_t*)alloc((size_t)NPIX * CHID * 2);
    float*  AXF   = (float*) alloc((size_t)NPIX * DD * 4);
    float*  UXF   = (float*) alloc((size_t)NPIX * DD * 4);
    bf16_t* AX16  = (bf16_t*)alloc((size_t)DD * NPAD * 2);
    float*  EMB1  = (float*) alloc((size_t)NPIX * DD * 4);
    float*  EMB2  = (float*) alloc((size_t)NPIX * DD * 4);
    float*  EMB1b = (float*) alloc((size_t)NPIX * DD * 4);
    float*  EMB2b = (float*) alloc((size_t)NPIX * DD * 4);
    bf16_t* EA1   = (bf16_t*)alloc((size_t)NPIX * DD * 2);
    bf16_t* EA2   = (bf16_t*)alloc((size_t)NPIX * DD * 2);
    bf16_t* EA3   = (bf16_t*)alloc((size_t)NPIX * DD * 2);
    bf16_t* EB1   = (bf16_t*)alloc((size_t)DD * NPAD * 2);
    bf16_t* W16   = (bf16_t*)alloc((size_t)CHID * KCONV * 2);
    bf16_t* AW0_16= (bf16_t*)alloc((size_t)DD * CHID * 2);
    bf16_t* UW0_16= (bf16_t*)alloc((size_t)DD * CHID * 2);
    bf16_t* AW1_16= (bf16_t*)alloc((size_t)DD * DD * 2);
    bf16_t* UW1_16= (bf16_t*)alloc((size_t)DD * DD * 2);
    bf16_t* SYM0  = (bf16_t*)alloc((size_t)DD * DD * 2);
    bf16_t* SYM1  = (bf16_t*)alloc((size_t)DD * DD * 2);
    bf16_t* CWT16 = (bf16_t*)alloc((size_t)DD * DD * 2);
    bf16_t* CWB16 = (bf16_t*)alloc((size_t)DD * DD * 2);
    char*   RegA  = alloc((size_t)2 * NPIX * NPIX * 4);   // AS | AQ ; R overlays AS
    char*   RegB  = alloc((size_t)NPIX * KCONV * 2);      // im2col ; later A16 pairs
    if (off > ws_size) return;

    float*  AS    = (float*)RegA;
    float*  AQ    = (float*)(RegA + (size_t)NPIX * NPIX * 4);
    float*  Rf    = AS;                                   // overlay (lifetimes disjoint)
    bf16_t* IM16  = (bf16_t*)RegB;
    bf16_t* A16s  = (bf16_t*)RegB;                                        // [NPIX][NPAD]
    bf16_t* A16q  = (bf16_t*)(RegB + (size_t)NPIX * NPAD * 2);
    bf16_t* RT16  = A16s;                                 // overlay (lifetimes disjoint)
    bf16_t* R16A  = A16q;

    // ---- launch helpers ----
    auto gemm = [&](const bf16_t* A, const bf16_t* B, float* C, const float* Ci,
                    const float* bias, int M, int Nc, int KP, int ldc, int bm, int relu) {
        int tiles = cdiv(M, 32) * cdiv(Nc, 64);
        k_gemm<<<cdiv(tiles, 8), 256, 0, stream>>>(A, B, C, Ci, bias, M, Nc, KP, ldc, bm, relu);
    };
    auto stageA = [&](const float* s, bf16_t* d, int M, int K, int lds, int KP) {
        k_stageA<<<cdiv(M * KP, 256), 256, 0, stream>>>(s, d, M, K, lds, KP);
    };
    auto stageB = [&](const float* s, bf16_t* d, int K, int Nc, int lds, int KP) {
        k_stageB<<<cdiv(Nc * KP, 256), 256, 0, stream>>>(s, d, K, Nc, lds, KP);
    };
    auto softmaxCol = [&](float* A) {
        k_softmax_col<<<cdiv(NPIX, 256), 256, 0, stream>>>(A, NPIX, NPIX);
    };
    auto minmaxNorm = [&](float* A, size_t n) {
        k_minmax_partial<<<256, 256, 0, stream>>>(A, n, RED);
        k_minmax_final<<<1, 256, 0, stream>>>(RED, 256);
        k_norm<<<2048, 256, 0, stream>>>(A, n, RED);
    };

    // ---- stage constant weights (bf16) ----
    stageA(conv_w, W16, CHID, KCONV, KCONV, KCONV);
    stageB(a_w0, AW0_16, CHID, DD, DD, CHID);
    stageB(u_w0, UW0_16, CHID, DD, DD, CHID);
    stageB(a_w1, AW1_16, DD, DD, DD, DD);
    stageB(u_w1, UW1_16, DD, DD, DD, DD);
    k_stageSymB<<<cdiv(DD * DD, 256), 256, 0, stream>>>(aff0, SYM0, DD);
    k_stageSymB<<<cdiv(DD * DD, 256), 256, 0, stream>>>(aff1, SYM1, DD);
    stageB(cross_w,                     CWT16, DD, DD, DD, DD);
    stageB(cross_w + (size_t)DD * DD,   CWB16, DD, DD, DD, DD);

    // ---- mask resize + conv (implicit GEMM, bias per output channel = row) ----
    k_resize_mask<<<cdiv(NPIX, 256), 256, 0, stream>>>(maskp, PM);
    for (int b = 0; b < 2; ++b) {
        k_im2colT<<<cdiv(NPIX * 1024, 256), 256, 0, stream>>>(x, PM, IM16, b);
        gemm(W16, IM16, FEAT + (size_t)b * CHID * NPIX, nullptr, conv_b,
             CHID, NPIX, KCONV, NPIX, /*bias row*/2, 0);
    }
    // FEAT[b] ([512][2704] row-major) reinterpreted as x_s/x_q [2704][512] (ref reshape)
    const float* xsF = FEAT;
    const float* xqF = FEAT + (size_t)CHID * NPIX;
    stageA(xsF, XS16, NPIX, CHID, CHID, CHID);
    stageA(xqF, XQ16, NPIX, CHID, CHID, CHID);

    // ---- round 1: Gram + column softmax (gconv col-normalization is identity) ----
    gemm(XS16, XS16, AS, nullptr, nullptr, NPIX, NPIX, CHID, NPIX, 0, 0);
    gemm(XQ16, XQ16, AQ, nullptr, nullptr, NPIX, NPIX, CHID, NPIX, 0, 0);
    softmaxCol(AS);
    softmaxCol(AQ);
    stageA(AS, A16s, NPIX, NPIX, NPIX, NPAD);
    stageA(AQ, A16q, NPIX, NPIX, NPIX, NPAD);

    // ---- gconv round 1:  emb = A @ relu(x@aw+ab) + relu(x@uw+ub) ----
    // s-path
    gemm(XS16, AW0_16, AXF, nullptr, a_b0, NPIX, DD, CHID, DD, 1, 1);
    gemm(XS16, UW0_16, UXF, nullptr, u_b0, NPIX, DD, CHID, DD, 1, 1);
    stageB(AXF, AX16, NPIX, DD, DD, NPAD);
    gemm(A16s, AX16, EMB1, UXF, nullptr, NPIX, DD, NPAD, DD, 0, 0);
    // q-path
    gemm(XQ16, AW0_16, AXF, nullptr, a_b0, NPIX, DD, CHID, DD, 1, 1);
    gemm(XQ16, UW0_16, UXF, nullptr, u_b0, NPIX, DD, CHID, DD, 1, 1);
    stageB(AXF, AX16, NPIX, DD, DD, NPAD);
    gemm(A16q, AX16, EMB2, UXF, nullptr, NPIX, DD, NPAD, DD, 0, 0);

    // ---- affinity 1: R = emb2 @ Asym0 @ emb1^T, then min-max normalize ----
    stageA(EMB2, EA3, NPIX, DD, DD, DD);
    gemm(EA3, SYM0, AXF, nullptr, nullptr, NPIX, DD, DD, DD, 0, 0);   // T1
    stageA(AXF, EA2, NPIX, DD, DD, DD);
    stageA(EMB1, EA1, NPIX, DD, DD, DD);                               // B-op of emb1^T
    gemm(EA2, EA1, Rf, nullptr, nullptr, NPIX, NPIX, DD, NPIX, 0, 0);
    minmaxNorm(Rf, (size_t)NPIX * NPIX);

    // ---- cross: e1 = emb1@Wt + (R^T@emb2)@Wb + b ; e2 = emb2@Wt + (R@emb1)@Wb + b ----
    stageB(Rf, RT16, NPIX, NPIX, NPIX, NPAD);   // A-op of R^T
    stageA(Rf, R16A, NPIX, NPIX, NPIX, NPAD);   // A-op of R
    stageB(EMB2, EB1, NPIX, DD, DD, NPAD);
    gemm(RT16, EB1, AXF, nullptr, nullptr, NPIX, DD, NPAD, DD, 0, 0);  // M1 = R^T@emb2
    stageA(EMB1, EA1, NPIX, DD, DD, DD);
    gemm(EA1, CWT16, EMB1b, nullptr, cross_b, NPIX, DD, DD, DD, 1, 0);
    stageA(AXF, EA2, NPIX, DD, DD, DD);
    gemm(EA2, CWB16, EMB1b, EMB1b, nullptr, NPIX, DD, DD, DD, 0, 0);
    stageB(EMB1, EB1, NPIX, DD, DD, NPAD);
    gemm(R16A, EB1, UXF, nullptr, nullptr, NPIX, DD, NPAD, DD, 0, 0);  // M2 = R@emb1
    stageA(EMB2, EA1, NPIX, DD, DD, DD);
    gemm(EA1, CWT16, EMB2b, nullptr, cross_b, NPIX, DD, DD, DD, 1, 0);
    stageA(UXF, EA2, NPIX, DD, DD, DD);
    gemm(EA2, CWB16, EMB2b, EMB2b, nullptr, NPIX, DD, DD, DD, 0, 0);

    // ---- round 2: Gram + softmax on new embeddings ----
    stageA(EMB1b, EA1, NPIX, DD, DD, DD);
    gemm(EA1, EA1, AS, nullptr, nullptr, NPIX, NPIX, DD, NPIX, 0, 0);
    softmaxCol(AS);
    stageA(AS, A16s, NPIX, NPIX, NPIX, NPAD);
    stageA(EMB2b, EA2, NPIX, DD, DD, DD);
    gemm(EA2, EA2, AQ, nullptr, nullptr, NPIX, NPIX, DD, NPIX, 0, 0);
    softmaxCol(AQ);
    stageA(AQ, A16q, NPIX, NPIX, NPIX, NPAD);

    // ---- gconv round 2 ----
    gemm(EA1, AW1_16, AXF, nullptr, a_b1, NPIX, DD, DD, DD, 1, 1);
    gemm(EA1, UW1_16, UXF, nullptr, u_b1, NPIX, DD, DD, DD, 1, 1);
    stageB(AXF, AX16, NPIX, DD, DD, NPAD);
    gemm(A16s, AX16, EMB1, UXF, nullptr, NPIX, DD, NPAD, DD, 0, 0);
    gemm(EA2, AW1_16, AXF, nullptr, a_b1, NPIX, DD, DD, DD, 1, 1);
    gemm(EA2, UW1_16, UXF, nullptr, u_b1, NPIX, DD, DD, DD, 1, 1);
    stageB(AXF, AX16, NPIX, DD, DD, NPAD);
    gemm(A16q, AX16, EMB2, UXF, nullptr, NPIX, DD, NPAD, DD, 0, 0);

    // ---- affinity 2 -> d_out, normalize in place ----
    float* Rout = (float*)d_out;
    stageA(EMB2, EA3, NPIX, DD, DD, DD);
    gemm(EA3, SYM1, AXF, nullptr, nullptr, NPIX, DD, DD, DD, 0, 0);
    stageA(AXF, EA2, NPIX, DD, DD, DD);
    stageA(EMB1, EA1, NPIX, DD, DD, DD);
    gemm(EA2, EA1, Rout, nullptr, nullptr, NPIX, NPIX, DD, NPIX, 0, 0);
    minmaxNorm(Rout, (size_t)NPIX * NPIX);

    // ---- second output: emb2 flat copy (ref reshape is a raw view) ----
    hipMemcpyAsync(Rout + (size_t)NPIX * NPIX, EMB2, (size_t)NPIX * DD * 4,
                   hipMemcpyDeviceToDevice, stream);
}
